// CrossAttentionLayer_7937099563619
// MI455X (gfx1250) — matrix operand order repaired and checked by reading, not compiled, not run on hardware
//
#include <hip/hip_runtime.h>

// ---------------------------------------------------------------------------
// CDNA5 (gfx1250) cross-attention, bf16 WMMA pipeline.
//   B=4, T=TC=1024, C=CTX_C=1024, H=16, HD=64
// Pipeline:
//   Q = x @ Wq, K = ctx @ Wk, V = ctx @ Wv    (fp32 in -> bf16 out, WMMA gemm)
//   flash attention per (b, h, 64-row q chunk):
//       double-buffered ASYNC global->LDS K/V tile staging (ASYNCcnt),
//       S = Q K^T via WMMA, mask + online softmax, O += P V via WMMA
//   out = O @ Wo                              (bf16 in -> fp32 out, WMMA gemm)
// Workspace: 4 * 4096*1024 bf16 = 32 MB.
// ---------------------------------------------------------------------------

typedef unsigned short u16t;
typedef unsigned int   u32t;
typedef __attribute__((ext_vector_type(16))) __bf16 v16bf;
typedef __attribute__((ext_vector_type(8)))  float  v8f;
typedef __attribute__((ext_vector_type(4)))  int    v4i;

union FragB { v16bf v; u32t u[8]; };
union FragF { v8f   v; float f[8]; };

// ---- CDNA5 async global->LDS copy (guarded; falls back to register copy) ---
#if defined(__has_builtin)
# if __has_builtin(__builtin_amdgcn_global_load_async_to_lds_b128)
#  define USE_ASYNC_LDS 1
# endif
# if __has_builtin(__builtin_amdgcn_s_wait_asynccnt)
#  define USE_WAIT_ASYNC 1
# endif
#endif

#if defined(USE_ASYNC_LDS)
typedef __attribute__((address_space(1))) v4i* gbl_v4i_p;   // global ptr (64-bit)
typedef __attribute__((address_space(3))) v4i* lds_v4i_p;   // LDS ptr (32-bit)
#endif

__device__ __forceinline__ void async_copy_b128(const u16t* g, u16t* l) {
#if defined(USE_ASYNC_LDS)
  // generic->AS1: identical 64-bit value; generic LDS ptr low 32 bits are the
  // LDS byte offset (ISA aperture rule), which is what an AS3 pointer holds.
  __builtin_amdgcn_global_load_async_to_lds_b128(
      (gbl_v4i_p)(unsigned long long)g,
      (lds_v4i_p)(unsigned int)(unsigned long long)l,
      0, 0);
#else
  const u32t* gs = (const u32t*)g;
  u32t x0 = gs[0], x1 = gs[1], x2 = gs[2], x3 = gs[3];
  u32t* ls = (u32t*)l;
  ls[0] = x0; ls[1] = x1; ls[2] = x2; ls[3] = x3;
#endif
}

#if defined(USE_WAIT_ASYNC)
# define WAIT_ASYNC_4() __builtin_amdgcn_s_wait_asynccnt(4)
# define WAIT_ASYNC_0() __builtin_amdgcn_s_wait_asynccnt(0)
#elif defined(USE_ASYNC_LDS)
# define WAIT_ASYNC_4() asm volatile("s_wait_asynccnt 0x4" ::: "memory")
# define WAIT_ASYNC_0() asm volatile("s_wait_asynccnt 0x0" ::: "memory")
#else
# define WAIT_ASYNC_4() ((void)0)
# define WAIT_ASYNC_0() ((void)0)
#endif

__device__ __forceinline__ u16t f2bf(float x) {
  union { float f; u32t u; } a; a.f = x;
  u32t r = a.u + 0x7FFFu + ((a.u >> 16) & 1u);   // round-to-nearest-even
  return (u16t)(r >> 16);
}

// A-matrix fragment (16x32 bf16), row-major source with given element stride.
// Layout: lane%16 = row M; lanes<16 hold K 0..15, lanes>=16 hold
// K 16..31; each VGPR = 2 contiguous K values (dword loads).
__device__ __forceinline__ v16bf load_a_frag(const u16t* base, int stride, int lane) {
  const int row  = lane & 15;
  const int kOff = (lane < 16) ? 0 : 16;
  const u16t* p = base + row * stride;
  FragB f;
#pragma unroll
  for (int j = 0; j < 4; ++j) f.u[j]     = *(const u32t*)(p + kOff + 2 * j);
#pragma unroll
  for (int j = 0; j < 4; ++j) f.u[4 + j] = *(const u32t*)(p + 8 + kOff + 2 * j);
  return f.v;
}

// B-matrix fragment (32x16 bf16) from storage where the WMMA-K dim is the
// *contiguous* index: buf[n][k], element stride `stride` between n rows.
// lane%16 = column N; lanes<16 hold K 0..15, lanes>=16 hold K 16..31.
__device__ __forceinline__ v16bf load_b_fragT(const u16t* base, int stride, int lane) {
  const int n    = lane & 15;
  const int kOff = (lane < 16) ? 0 : 16;
  const u16t* p = base + n * stride + kOff;
  FragB f;
#pragma unroll
  for (int j = 0; j < 8; ++j) f.u[j] = *(const u32t*)(p + 2 * j);
  return f.v;
}

// B-matrix fragment (32x16 bf16) from row-major storage buf[k][n] (WMMA-K dim
// is the row index): gathers two u16 per VGPR and packs.
__device__ __forceinline__ v16bf load_b_fragRM(const u16t* base, int stride,
                                               int nBase, int lane) {
  const int n    = nBase + (lane & 15);
  const int kOff = (lane < 16) ? 0 : 16;
  FragB f;
#pragma unroll
  for (int j = 0; j < 8; ++j) {
    u32t lo = base[(kOff + 2 * j) * stride + n];
    u32t hi = base[(kOff + 2 * j + 1) * stride + n];
    f.u[j] = lo | (hi << 16);
  }
  return f.v;
}

// ---------------------------------------------------------------------------
// Tiled GEMM: Out[M,N] = A[M,K] @ Bw[K,N].
// A is fp32 (converted) or bf16; Bw is fp32 (converted); Out bf16 or fp32.
// Block tile 128x64, 8 waves (256 thr), each wave a 32x32 tile (2x2 WMMA).
// Next k-tile is prefetched (global_prefetch) while current tile computes.
// ---------------------------------------------------------------------------
template <bool A_F32, bool OUT_F32>
__global__ __launch_bounds__(256) void gemm_wmma_bf16(
    const void* __restrict__ Ap, const float* __restrict__ Bw,
    void* __restrict__ Out, int M, int N, int K) {
  constexpr int BM = 128, BN = 64, BK = 32;
  constexpr int AS = BK + 2;  // 34, padded LDS stride (bank-conflict break)
  constexpr int BS = BK + 4;  // 36
  __shared__ __align__(16) u16t As[BM * AS];
  __shared__ __align__(16) u16t Bt[BN * BS];   // transposed: Bt[n][k]

  const int tid  = threadIdx.x;
  const int lane = tid & 31;
  const int wave = tid >> 5;
  const int wm = wave & 3, wn = wave >> 2;
  const int m0 = blockIdx.y * BM;
  const int n0 = blockIdx.x * BN;

  const float* Af = (const float*)Ap;
  const u16t*  Ab = (const u16t*)Ap;

  FragF acc[2][2];
#pragma unroll
  for (int i = 0; i < 2; ++i)
#pragma unroll
    for (int j = 0; j < 2; ++j)
#pragma unroll
      for (int r = 0; r < 8; ++r) acc[i][j].f[r] = 0.0f;

  for (int k0 = 0; k0 < K; k0 += BK) {
    // Stage A tile (BM x BK) as bf16.
#pragma unroll
    for (int i = 0; i < (BM * BK) / 256; ++i) {
      int e = tid + i * 256;
      int r = e >> 5, c = e & 31;
      u16t b;
      if (A_F32) b = f2bf(Af[(size_t)(m0 + r) * K + k0 + c]);
      else       b = Ab[(size_t)(m0 + r) * K + k0 + c];
      As[r * AS + c] = b;
    }
    // Stage B tile (BK x BN) transposed -> Bt[n][k], bf16.
#pragma unroll
    for (int i = 0; i < (BK * BN) / 256; ++i) {
      int e = tid + i * 256;
      int r = e >> 6, c = e & 63;
      Bt[c * BS + r] = f2bf(Bw[(size_t)(k0 + r) * N + n0 + c]);
    }
    // Prefetch next k-tile into cache while this tile is consumed.
    if (k0 + BK < K) {
      int ra = tid >> 1, ca = (tid & 1) * 16;
      if (A_F32) __builtin_prefetch(&Af[(size_t)(m0 + ra) * K + k0 + BK + ca], 0, 0);
      else       __builtin_prefetch(&Ab[(size_t)(m0 + ra) * K + k0 + BK + ca], 0, 0);
      int rb = tid >> 3, cb = (tid & 7) * 8;
      __builtin_prefetch(&Bw[(size_t)(k0 + BK + rb) * N + n0 + cb], 0, 0);
    }
    __syncthreads();

    v16bf a[2], bfr[2];
#pragma unroll
    for (int mt = 0; mt < 2; ++mt)
      a[mt] = load_a_frag(&As[(wm * 32 + mt * 16) * AS], AS, lane);
#pragma unroll
    for (int nt = 0; nt < 2; ++nt)
      bfr[nt] = load_b_fragT(&Bt[(wn * 32 + nt * 16) * BS], BS, lane);
#pragma unroll
    for (int mt = 0; mt < 2; ++mt)
#pragma unroll
      for (int nt = 0; nt < 2; ++nt)
        acc[mt][nt].v = __builtin_amdgcn_wmma_f32_16x16x32_bf16(
            false, a[mt], false, bfr[nt], (short)0, acc[mt][nt].v, false, false);
    __syncthreads();
  }

  float* Of = (float*)Out;
  u16t*  Ob = (u16t*)Out;
#pragma unroll
  for (int mt = 0; mt < 2; ++mt)
#pragma unroll
    for (int nt = 0; nt < 2; ++nt)
#pragma unroll
      for (int r = 0; r < 8; ++r) {
        int row = m0 + wm * 32 + mt * 16 + ((lane < 16) ? r : 8 + r);
        int col = n0 + wn * 32 + nt * 16 + (lane & 15);
        float v = acc[mt][nt].f[r];
        if (OUT_F32) Of[(size_t)row * N + col] = v;
        else         Ob[(size_t)row * N + col] = f2bf(v);
      }
}

// ---------------------------------------------------------------------------
// Flash attention: one block = 64 q rows of one (b, h); 4 waves, 16 rows each.
// Double-buffered ASYNC staging of 32-key K/V tiles (row-major, [s][d],
// element stride 72 -> 144B rows keep every B128 chunk 16B-aligned).
// Per tile: S = Q K^T (4 WMMA), mask + online softmax, O += P V (4 WMMA).
// ---------------------------------------------------------------------------
__global__ __launch_bounds__(128) void attn_wmma(
    const u16t* __restrict__ Qb, const u16t* __restrict__ Kb,
    const u16t* __restrict__ Vb, const int* __restrict__ Mk,
    u16t* __restrict__ Ob) {
  constexpr int T = 1024, TC = 1024, C = 1024, HD = 64;
  constexpr int SR = 72;   // padded tile stride; 144B = 9*16B per row
  constexpr int PS = 36;
  __shared__ __align__(16) u16t Ks[2][32 * SR];   // K tiles [s][d]
  __shared__ __align__(16) u16t Vs[2][32 * SR];   // V tiles [s][d]
  __shared__ __align__(16) u16t Pb[4 * 16 * PS];  // per-wave P scratch [m][s]

  const int tid = threadIdx.x, lane = tid & 31, wave = tid >> 5;
  const int b = blockIdx.z, h = blockIdx.y;
  const int q0 = blockIdx.x * 64 + wave * 16;

  const u16t* gK = Kb + ((size_t)(b * TC) * C + h * HD);
  const u16t* gV = Vb + ((size_t)(b * TC) * C + h * HD);

  // Q fragments: 16x64 as two 16x32 A-frags, straight from global bf16.
  v16bf qa[2];
#pragma unroll
  for (int cc = 0; cc < 2; ++cc)
    qa[cc] = load_a_frag(Qb + ((size_t)(b * T + q0) * C + h * HD + cc * 32), C, lane);

  FragF o[4];
  float m8[8], l8[8];
#pragma unroll
  for (int j = 0; j < 4; ++j)
#pragma unroll
    for (int r = 0; r < 8; ++r) o[j].f[r] = 0.0f;
#pragma unroll
  for (int r = 0; r < 8; ++r) { m8[r] = -3.0e38f; l8[r] = 0.0f; }

  const float scale = 0.03125f;  // 1/sqrt(C)
  constexpr int NIT = TC / 32;   // 32 key tiles

  // Issue async staging of tile 0: each thread moves 2 B128 chunks of K and V.
#pragma unroll
  for (int i = 0; i < 2; ++i) {
    int chunk = tid + i * 128;              // 0..255
    int r = chunk >> 3, c8 = (chunk & 7) << 3;
    async_copy_b128(gK + (size_t)r * C + c8, &Ks[0][r * SR + c8]);
    async_copy_b128(gV + (size_t)r * C + c8, &Vs[0][r * SR + c8]);
  }

  for (int it = 0; it < NIT; ++it) {
    const int cur = it & 1;
    const int sBase = it * 32;
    // Issue tile it+1 into the other buffer (its readers finished at the
    // barrier closing iteration it-1).
    if (it + 1 < NIT) {
#pragma unroll
      for (int i = 0; i < 2; ++i) {
        int chunk = tid + i * 128;
        int r = chunk >> 3, c8 = (chunk & 7) << 3;
        size_t g = (size_t)(sBase + 32 + r) * C + c8;
        async_copy_b128(gK + g, &Ks[cur ^ 1][r * SR + c8]);
        async_copy_b128(gV + g, &Vs[cur ^ 1][r * SR + c8]);
      }
      WAIT_ASYNC_4();   // tile `it` (my 4 older transfers) complete, in order
    } else {
      WAIT_ASYNC_0();
    }
    __syncthreads();

    const u16t* K0 = Ks[cur];
    const u16t* V0 = Vs[cur];

    // S = Q @ K^T  (16 rows x 32 keys = two 16x16 accumulators)
    FragF s0, s1;
#pragma unroll
    for (int r = 0; r < 8; ++r) { s0.f[r] = 0.0f; s1.f[r] = 0.0f; }
#pragma unroll
    for (int cc = 0; cc < 2; ++cc) {
      v16bf kb0 = load_b_fragT(K0 + cc * 32, SR, lane);            // keys 0..15
      v16bf kb1 = load_b_fragT(K0 + 16 * SR + cc * 32, SR, lane);  // keys 16..31
      s0.v = __builtin_amdgcn_wmma_f32_16x16x32_bf16(false, qa[cc], false, kb0,
                                                     (short)0, s0.v, false, false);
      s1.v = __builtin_amdgcn_wmma_f32_16x16x32_bf16(false, qa[cc], false, kb1,
                                                     (short)0, s1.v, false, false);
    }

    // Mask, online softmax, write P (bf16) to per-wave LDS scratch.
    u16t* Pw = &Pb[wave * 16 * PS];
#pragma unroll
    for (int r = 0; r < 8; ++r) {
      const int rloc = (lane < 16) ? r : 8 + r;      // row within 16-row tile
      const int trow = q0 + rloc;
      const int* mrow = Mk + (size_t)(b * T + trow) * TC + sBase;
      const int c0 = lane & 15, c1 = c0 + 16;
      float x0 = s0.f[r] * scale; if (mrow[c0] == 0) x0 = -10000.0f;
      float x1 = s1.f[r] * scale; if (mrow[c1] == 0) x1 = -10000.0f;
      float mx = fmaxf(x0, x1);
#pragma unroll
      for (int off = 1; off < 16; off <<= 1) mx = fmaxf(mx, __shfl_xor(mx, off));
      float mnew  = fmaxf(m8[r], mx);
      float alpha = __expf(m8[r] - mnew);
      float p0 = __expf(x0 - mnew);
      float p1 = __expf(x1 - mnew);
      float rs = p0 + p1;
#pragma unroll
      for (int off = 1; off < 16; off <<= 1) rs += __shfl_xor(rs, off);
      l8[r] = l8[r] * alpha + rs;
      m8[r] = mnew;
#pragma unroll
      for (int j = 0; j < 4; ++j) o[j].f[r] *= alpha;
      Pw[rloc * PS + c0] = f2bf(p0);
      Pw[rloc * PS + c1] = f2bf(p1);
    }
    __syncthreads();

    // O += P @ V : P is 16x32 A-frag; V tiles are 32x16 B-frags from
    // row-major V (K dim = key row -> packed u16 gathers).
    v16bf pa = load_a_frag(Pw, PS, lane);
#pragma unroll
    for (int j = 0; j < 4; ++j) {
      v16bf vb = load_b_fragRM(V0, SR, j * 16, lane);
      o[j].v = __builtin_amdgcn_wmma_f32_16x16x32_bf16(false, pa, false, vb,
                                                       (short)0, o[j].v, false, false);
    }
    __syncthreads();
  }

  // Normalize and write O (bf16) back in [B, T, C] head-concat layout.
#pragma unroll
  for (int j = 0; j < 4; ++j)
#pragma unroll
    for (int r = 0; r < 8; ++r) {
      int trow = q0 + ((lane < 16) ? r : 8 + r);
      int d = j * 16 + (lane & 15);
      float inv = 1.0f / (l8[r] + 1e-20f);
      Ob[(size_t)(b * T + trow) * C + h * HD + d] = f2bf(o[j].f[r] * inv);
    }
}

// ---------------------------------------------------------------------------
extern "C" void kernel_launch(void* const* d_in, const int* in_sizes, int n_in,
                              void* d_out, int out_size, void* d_ws, size_t ws_size,
                              hipStream_t stream) {
  (void)in_sizes; (void)n_in; (void)out_size; (void)ws_size;
  const float* x   = (const float*)d_in[0];
  const float* ctx = (const float*)d_in[1];
  const int*   msk = (const int*)d_in[2];
  const float* Wq  = (const float*)d_in[3];
  const float* Wk  = (const float*)d_in[4];
  const float* Wv  = (const float*)d_in[5];
  const float* Wo  = (const float*)d_in[6];

  const int Bn = 4, T = 1024, TC = 1024, C = 1024;
  const size_t elems = (size_t)Bn * T * C;
  u16t* Qb = (u16t*)d_ws;      // bf16 workspace buffers, 8 MB each
  u16t* Kb = Qb + elems;
  u16t* Vb = Kb + elems;
  u16t* Ob = Vb + elems;

  dim3 blk(256), grd(C / 64, (Bn * T) / 128);
  gemm_wmma_bf16<true,  false><<<grd, blk, 0, stream>>>(x,   Wq, Qb, Bn * T,  C, C);
  gemm_wmma_bf16<true,  false><<<grd, blk, 0, stream>>>(ctx, Wk, Kb, Bn * TC, C, C);
  gemm_wmma_bf16<true,  false><<<grd, blk, 0, stream>>>(ctx, Wv, Vb, Bn * TC, C, C);
  attn_wmma<<<dim3(T / 64, 16, Bn), 128, 0, stream>>>(Qb, Kb, Vb, msk, Ob);
  gemm_wmma_bf16<false, true ><<<grd, blk, 0, stream>>>(Ob, Wo, d_out, Bn * T, C, C);
}
